// GCN_mhealth_Attn_29944511988453
// MI455X (gfx1250) — compile-verified, matrix operand order checked by reading, not run on hardware
//
#include <hip/hip_runtime.h>
#include <hip/hip_bf16.h>
#include <stdint.h>

#define F_IN   128
#define HC     256
#define NH     4
#define HID    128
#define NCLS   12
#define NEGSL  0.2f
#define BN_EPS 1e-5f

typedef __attribute__((ext_vector_type(16))) __bf16 v16bf;
typedef __attribute__((ext_vector_type(8)))  __bf16 v8bf;
typedef __attribute__((ext_vector_type(8)))  float  v8f;
typedef __attribute__((ext_vector_type(4)))  unsigned v4u;
typedef __attribute__((ext_vector_type(8)))  int      v8i;
typedef __attribute__((ext_vector_type(4)))  int      v4i;

// ---- monotonic float<->uint keys so segment-max can use integer atomicMax ----
static __device__ __forceinline__ unsigned fkey(float f) {
  unsigned u = __float_as_uint(f);
  return (u & 0x80000000u) ? ~u : (u | 0x80000000u);
}
static __device__ __forceinline__ float funkey(unsigned k) {
  unsigned u = (k & 0x80000000u) ? (k & 0x7fffffffu) : ~k;
  return __uint_as_float(u);
}
static __device__ __forceinline__ float lrelu(float v) { return v > 0.f ? v : NEGSL * v; }

// ---------------------------------------------------------------------------
// fp32 -> bf16 elementwise convert
// ---------------------------------------------------------------------------
__global__ void f32_to_bf16_kernel(const float* __restrict__ in, __bf16* __restrict__ out, size_t n) {
  size_t i = (size_t)blockIdx.x * blockDim.x + threadIdx.x;
  if (i < n) out[i] = (__bf16)in[i];
}

// ---------------------------------------------------------------------------
// Pack a row-major fp32 [K,N] matrix into WMMA B-fragment order (bf16).
// Fragment f=(kt,nt): 32 lanes x 16 bf16, contiguous per lane.
// Lane l<16 holds column nt*16+l, k in {0..7, 16..23}; lane l+16 holds k in {8..15, 24..31}.
// ---------------------------------------------------------------------------
__global__ void pack_b_kernel(const float* __restrict__ W, __bf16* __restrict__ Bpk, int K, int N) {
  int idx = blockIdx.x * blockDim.x + threadIdx.x;
  if (idx >= K * N) return;
  int i    = idx & 15;
  int lane = (idx >> 4) & 31;
  int f    = idx >> 9;
  int ntiles = N >> 4;
  int kt = f / ntiles, nt = f % ntiles;
  int col = nt * 16 + (lane & 15);
  int kin = (lane >> 4) << 3;                       // 0 or 8
  int k   = kt * 32 + ((i < 8) ? (kin + i) : (kin + 16 + (i - 8)));
  Bpk[idx] = (__bf16)W[(size_t)k * N + col];
}

// ---------------------------------------------------------------------------
// Generic WMMA GEMM: C[M,N] = A[M,K](bf16, row-major) * Bpk (fragment-packed)
// Block = 128 threads (4 waves); wave owns a 16x64 tile; grid.y tiles N by 64.
// The block's B panel (nkt rows x 4 fragments x 1KB) is staged into LDS once
// via the Tensor Data Mover (2D D# descriptor), then WMMA reads ds_load_b128.
// ---------------------------------------------------------------------------
__global__ void __launch_bounds__(128) wmma_gemm_kernel(
    const __bf16* __restrict__ A, const __bf16* __restrict__ Bpk,
    float* __restrict__ C, int M, int K, int N) {
  __shared__ __bf16 smemB[16384];                    // 32KB: up to 8 kt-rows x 4 frags x 512
  const int lane = threadIdx.x & 31;
  const int wave = threadIdx.x >> 5;
  const int n0     = blockIdx.y * 64;
  const int ntiles = N >> 4;
  const int nkt    = K >> 5;

  // ---- stage B panel into LDS via Tensor Data Mover ----
  if (wave == 0) {
    // D# group 0: count=1 | lds_addr | global_addr(57b) | type=2
    unsigned lds_base = (unsigned)(uintptr_t)(&smemB[0]);          // LDS offset = flat[31:0]
    uint64_t ga = (uint64_t)(uintptr_t)Bpk + (uint64_t)(n0 >> 4) * 1024u; // panel start (bytes)
    v4u g0;
    g0[0] = 1u;                                                    // count=1, user mode
    g0[1] = lds_base;
    g0[2] = (unsigned)(ga & 0xFFFFFFFFu);
    g0[3] = (unsigned)((ga >> 32) & 0x01FFFFFFu) | (2u << 30);     // type=2 ("image")
    // D# group 1: data_size=4B; tile = 1024 dwords x nkt rows; row stride = ntiles*256 dwords
    unsigned td0  = 1024u;                         // tensor_dim0 (dwords)
    unsigned nrow = (unsigned)nkt;                 // tensor_dim1 / tile_dim1
    uint64_t s0   = (uint64_t)ntiles * 256u;       // tensor_dim0_stride (dwords)
    v8i g1;
    g1[0] = (int)(2u << 16);                                       // data_size=2 (4 bytes)
    g1[1] = (int)((td0 & 0xFFFFu) << 16);                          // tensor_dim0[15:0]
    g1[2] = (int)((td0 >> 16) | ((nrow & 0xFFFFu) << 16));         // td0[31:16] | td1[15:0]
    g1[3] = (int)((nrow >> 16) | (1024u << 16));                   // td1[31:16] | tile_dim0
    g1[4] = (int)(nrow & 0xFFFFu);                                 // tile_dim1 (tile_dim2=0)
    g1[5] = (int)(s0 & 0xFFFFFFFFu);                               // stride0[31:0]
    g1[6] = (int)((s0 >> 32) & 0xFFFFu);                           // stride0[47:32]
    g1[7] = 0;
    v4i z4 = {0, 0, 0, 0};
    v8i z8 = {0, 0, 0, 0, 0, 0, 0, 0};
    __builtin_amdgcn_tensor_load_to_lds(g0, g1, z4, z4, z8, 0);
    __builtin_amdgcn_s_wait_tensorcnt(0);
  }
  __syncthreads();

  const int m0 = blockIdx.x * 64 + wave * 16;
  if (m0 < M) {                                    // wave-uniform (M % 16 == 0)
    const int row  = m0 + (lane & 15);
    const int koff = (lane >> 4) << 3;             // hi half-lanes take k+8
    v8f acc0 = {}, acc1 = {}, acc2 = {}, acc3 = {};
    for (int kt = 0; kt < nkt; ++kt) {
      // A fragment: 16x32 bf16, ISA layout for 16-bit A (two 16B chunks per lane)
      const __bf16* ap = A + (size_t)row * K + kt * 32 + koff;
      v8bf lo = *(const v8bf*)(ap);
      v8bf hi = *(const v8bf*)(ap + 16);
      v16bf a;
#pragma unroll
      for (int i = 0; i < 8; ++i) { a[i] = lo[i]; a[i + 8] = hi[i]; }

      const __bf16* bp = smemB + ((size_t)kt * 4) * 512 + lane * 16;
      v16bf b0 = *(const v16bf*)(bp);
      v16bf b1 = *(const v16bf*)(bp + 512);
      v16bf b2 = *(const v16bf*)(bp + 1024);
      v16bf b3 = *(const v16bf*)(bp + 1536);

      acc0 = __builtin_amdgcn_wmma_f32_16x16x32_bf16(false, a, false, b0, (short)0, acc0, false, false);
      acc1 = __builtin_amdgcn_wmma_f32_16x16x32_bf16(false, a, false, b1, (short)0, acc1, false, false);
      acc2 = __builtin_amdgcn_wmma_f32_16x16x32_bf16(false, a, false, b2, (short)0, acc2, false, false);
      acc3 = __builtin_amdgcn_wmma_f32_16x16x32_bf16(false, a, false, b3, (short)0, acc3, false, false);
    }

    // D layout: lane<16 -> rows m0..m0+7, lane>=16 -> rows m0+8..m0+15
    float* cp = C + (size_t)(m0 + ((lane >> 4) << 3)) * N + n0 + (lane & 15);
#pragma unroll
    for (int j = 0; j < 8; ++j) {
      cp[(size_t)j * N +  0] = acc0[j];
      cp[(size_t)j * N + 16] = acc1[j];
      cp[(size_t)j * N + 32] = acc2[j];
      cp[(size_t)j * N + 48] = acc3[j];
    }
  }
}

// ---------------------------------------------------------------------------
// Per-node attention dots + self-loop logits + segment-max init (wave per node)
// ---------------------------------------------------------------------------
__global__ void node_attn_kernel(const float* __restrict__ h,
                                 const float* __restrict__ att_src,
                                 const float* __restrict__ att_dst,
                                 float* __restrict__ asrc, float* __restrict__ adst,
                                 float* __restrict__ eself, unsigned* __restrict__ emax,
                                 int N) {
  int lane = threadIdx.x & 31;
  int node = blockIdx.x * (blockDim.x >> 5) + (threadIdx.x >> 5);
  if (node >= N) return;
  int j0 = lane * 8, head = lane >> 3;
  const float* hp = h + (size_t)node * HC + j0;
  float ss = 0.f, sd = 0.f;
#pragma unroll
  for (int i = 0; i < 8; ++i) {
    float v = hp[i];
    ss += v * att_src[j0 + i];
    sd += v * att_dst[j0 + i];
  }
#pragma unroll
  for (int off = 1; off < 8; off <<= 1) {
    ss += __shfl_xor(ss, off, 32);
    sd += __shfl_xor(sd, off, 32);
  }
  if ((lane & 7) == 0) {
    asrc[node * NH + head] = ss;
    adst[node * NH + head] = sd;
    float e = lrelu(ss + sd);           // self-loop logit
    eself[node * NH + head] = e;
    emax[node * NH + head]  = fkey(e);  // init running max with self-loop
  }
}

// ---------------------------------------------------------------------------
// Edge pass 1: segment max via integer atomicMax on monotonic keys
// ---------------------------------------------------------------------------
__global__ void edge_max_kernel(const int* __restrict__ ei,
                                const float* __restrict__ asrc, const float* __restrict__ adst,
                                unsigned* __restrict__ emax, int E) {
  int e = blockIdx.x * blockDim.x + threadIdx.x;
  if (e >= E) return;
  int s = ei[e], d = ei[E + e];
#pragma unroll
  for (int hh = 0; hh < NH; ++hh) {
    float v = lrelu(asrc[s * NH + hh] + adst[d * NH + hh]);
    atomicMax(&emax[d * NH + hh], fkey(v));
  }
}

// ---------------------------------------------------------------------------
// Per-(node,head): denom = exp(self - max)
// ---------------------------------------------------------------------------
__global__ void node_denom_kernel(const float* __restrict__ eself,
                                  const unsigned* __restrict__ emax,
                                  float* __restrict__ denom, int n) {
  int i = blockIdx.x * blockDim.x + threadIdx.x;
  if (i < n) denom[i] = __expf(eself[i] - funkey(emax[i]));
}

// ---------------------------------------------------------------------------
// Edge pass 2: exp-sum; prefetch h[src] to warm L2 for the aggregation pass
// ---------------------------------------------------------------------------
__global__ void edge_expsum_kernel(const int* __restrict__ ei,
                                   const float* __restrict__ asrc, const float* __restrict__ adst,
                                   const unsigned* __restrict__ emax, float* __restrict__ denom,
                                   const float* __restrict__ h, int E) {
  int e = blockIdx.x * blockDim.x + threadIdx.x;
  if (e >= E) return;
  int s = ei[e], d = ei[E + e];
  __builtin_prefetch(h + (size_t)s * HC, 0, 0);   // global_prefetch_b8
#pragma unroll
  for (int hh = 0; hh < NH; ++hh) {
    float v = lrelu(asrc[s * NH + hh] + adst[d * NH + hh]);
    atomicAdd(&denom[d * NH + hh], __expf(v - funkey(emax[d * NH + hh])));
  }
}

// ---------------------------------------------------------------------------
// Edge pass 3: out[dst] += alpha * h[src]  (wave per edge; 8 f32 atomics/lane)
// ---------------------------------------------------------------------------
__global__ void edge_aggr_kernel(const int* __restrict__ ei,
                                 const float* __restrict__ asrc, const float* __restrict__ adst,
                                 const unsigned* __restrict__ emax, const float* __restrict__ denom,
                                 const float* __restrict__ h, float* __restrict__ out, int E) {
  int lane = threadIdx.x & 31;
  int e = blockIdx.x * (blockDim.x >> 5) + (threadIdx.x >> 5);
  if (e >= E) return;
  int s = ei[e], d = ei[E + e];
  int head = lane >> 3;
  float v = lrelu(asrc[s * NH + head] + adst[d * NH + head]);
  float alpha = __expf(v - funkey(emax[d * NH + head])) / denom[d * NH + head];
  int j0 = lane * 8;
  const float4* hp = (const float4*)(h + (size_t)s * HC + j0);
  float4 h0 = hp[0], h1 = hp[1];
  float* op = out + (size_t)d * HC + j0;
  atomicAdd(op + 0, alpha * h0.x); atomicAdd(op + 1, alpha * h0.y);
  atomicAdd(op + 2, alpha * h0.z); atomicAdd(op + 3, alpha * h0.w);
  atomicAdd(op + 4, alpha * h1.x); atomicAdd(op + 5, alpha * h1.y);
  atomicAdd(op + 6, alpha * h1.z); atomicAdd(op + 7, alpha * h1.w);
}

// ---------------------------------------------------------------------------
// Self-loop contribution + gat bias + ReLU + BatchNorm -> bf16 (wave per node)
// ---------------------------------------------------------------------------
__global__ void node_bn_kernel(const float* __restrict__ gat, const float* __restrict__ h,
                               const float* __restrict__ eself, const unsigned* __restrict__ emax,
                               const float* __restrict__ denom,
                               const float* __restrict__ gat_bias,
                               const float* __restrict__ g, const float* __restrict__ b,
                               const float* __restrict__ mean, const float* __restrict__ var,
                               __bf16* __restrict__ out, int N) {
  int lane = threadIdx.x & 31;
  int node = blockIdx.x * (blockDim.x >> 5) + (threadIdx.x >> 5);
  if (node >= N) return;
  int j0 = lane * 8, head = lane >> 3;
  int ih = node * NH + head;
  float alpha = __expf(eself[ih] - funkey(emax[ih])) / denom[ih];
  v8bf r;
#pragma unroll
  for (int i = 0; i < 8; ++i) {
    int j = j0 + i;
    float v = gat[(size_t)node * HC + j] + alpha * h[(size_t)node * HC + j] + gat_bias[j];
    v = fmaxf(v, 0.f);                                  // relu
    float sc = g[j] * rsqrtf(var[j] + BN_EPS);
    v = (v - mean[j]) * sc + b[j];
    r[i] = (__bf16)v;
  }
  *(v8bf*)(out + (size_t)node * HC + j0) = r;
}

// ---------------------------------------------------------------------------
// fc2 epilogue: bias + ReLU -> bf16
// ---------------------------------------------------------------------------
__global__ void bias_relu_bf16_kernel(const float* __restrict__ in, const float* __restrict__ bias,
                                      __bf16* __restrict__ out, size_t n) {
  size_t i = (size_t)blockIdx.x * blockDim.x + threadIdx.x;
  if (i >= n) return;
  float v = in[i] + bias[i & (HID - 1)];
  out[i] = (__bf16)fmaxf(v, 0.f);
}

// ---------------------------------------------------------------------------
// Final 128 -> 12 head (tiny; VALU). 16 threads per node, 12 active.
// ---------------------------------------------------------------------------
__global__ void final_kernel(const __bf16* __restrict__ hid, const float* __restrict__ out_w,
                             const float* __restrict__ out_b, float* __restrict__ out, int N) {
  int t = blockIdx.x * blockDim.x + threadIdx.x;
  int node = t >> 4, k = t & 15;
  if (node >= N || k >= NCLS) return;
  const __bf16* hp = hid + (size_t)node * HID;
  float acc = out_b[k];
#pragma unroll 4
  for (int j = 0; j < HID; ++j) acc += (float)hp[j] * out_w[j * NCLS + k];
  out[node * NCLS + k] = acc;
}

// ---------------------------------------------------------------------------
extern "C" void kernel_launch(void* const* d_in, const int* in_sizes, int n_in,
                              void* d_out, int out_size, void* d_ws, size_t ws_size,
                              hipStream_t stream) {
  (void)n_in; (void)out_size; (void)ws_size;
  const float* x        = (const float*)d_in[0];
  const int*   ei       = (const int*)d_in[1];
  const float* W        = (const float*)d_in[2];
  const float* att_src  = (const float*)d_in[3];
  const float* att_dst  = (const float*)d_in[4];
  const float* gat_bias = (const float*)d_in[5];
  const float* bn_gamma = (const float*)d_in[6];
  const float* bn_beta  = (const float*)d_in[7];
  const float* bn_mean  = (const float*)d_in[8];
  const float* bn_var   = (const float*)d_in[9];
  const float* fc2_w    = (const float*)d_in[10];
  const float* fc2_b    = (const float*)d_in[11];
  const float* out_w    = (const float*)d_in[12];
  const float* out_b    = (const float*)d_in[13];
  float* out = (float*)d_out;

  const int Nn = in_sizes[0] / F_IN;   // 50000
  const int E  = in_sizes[1] / 2;      // 800000

  // ---- workspace carve (256B aligned) ----
  char* p = (char*)d_ws;
  auto carve = [&](size_t bytes) { char* r = p; p += (bytes + 255) & ~(size_t)255; return r; };
  __bf16*   xbf   = (__bf16*)carve((size_t)Nn * F_IN * 2);   // reused as hid_bf16 after GEMM1
  __bf16*   Wpk   = (__bf16*)carve((size_t)F_IN * HC * 2);
  __bf16*   F2pk  = (__bf16*)carve((size_t)HC * HID * 2);
  float*    h     = (float*)carve((size_t)Nn * HC * 4);      // reused as fc2 out after node_bn
  float*    gat   = (float*)carve((size_t)Nn * HC * 4);
  float*    asrc  = (float*)carve((size_t)Nn * NH * 4);
  float*    adst  = (float*)carve((size_t)Nn * NH * 4);
  float*    eself = (float*)carve((size_t)Nn * NH * 4);
  unsigned* emax  = (unsigned*)carve((size_t)Nn * NH * 4);
  float*    denom = (float*)carve((size_t)Nn * NH * 4);
  __bf16*   h2bf  = (__bf16*)carve((size_t)Nn * HC * 2);

  (void)hipMemsetAsync(gat, 0, (size_t)Nn * HC * 4, stream);

  // convert + pack
  {
    size_t n = (size_t)Nn * F_IN;
    f32_to_bf16_kernel<<<(unsigned)((n + 255) / 256), 256, 0, stream>>>(x, xbf, n);
  }
  pack_b_kernel<<<(F_IN * HC + 255) / 256, 256, 0, stream>>>(W, Wpk, F_IN, HC);
  pack_b_kernel<<<(HC * HID + 255) / 256, 256, 0, stream>>>(fc2_w, F2pk, HC, HID);

  // GEMM1: h = x @ W   [Nn,128]x[128,256]
  wmma_gemm_kernel<<<dim3((Nn + 63) / 64, HC / 64), 128, 0, stream>>>(xbf, Wpk, h, Nn, F_IN, HC);

  // attention
  node_attn_kernel<<<(Nn + 7) / 8, 256, 0, stream>>>(h, att_src, att_dst, asrc, adst, eself, emax, Nn);
  edge_max_kernel<<<(E + 255) / 256, 256, 0, stream>>>(ei, asrc, adst, emax, E);
  node_denom_kernel<<<(Nn * NH + 255) / 256, 256, 0, stream>>>(eself, emax, denom, Nn * NH);
  edge_expsum_kernel<<<(E + 255) / 256, 256, 0, stream>>>(ei, asrc, adst, emax, denom, h, E);
  edge_aggr_kernel<<<(E + 7) / 8, 256, 0, stream>>>(ei, asrc, adst, emax, denom, h, gat, E);

  // self-loop + bias + relu + BN -> bf16
  node_bn_kernel<<<(Nn + 7) / 8, 256, 0, stream>>>(gat, h, eself, emax, denom, gat_bias,
                                                   bn_gamma, bn_beta, bn_mean, bn_var, h2bf, Nn);

  // GEMM2: fc2out = h2 @ fc2_w   [Nn,256]x[256,128]  (output reuses h buffer)
  wmma_gemm_kernel<<<dim3((Nn + 63) / 64, HID / 64), 128, 0, stream>>>(h2bf, F2pk, h, Nn, HC, HID);

  // bias + relu -> bf16 (reuses xbf buffer)
  {
    size_t n = (size_t)Nn * HID;
    bias_relu_bf16_kernel<<<(unsigned)((n + 255) / 256), 256, 0, stream>>>(h, fc2_b, xbf, n);
  }

  // final head
  final_kernel<<<(Nn * 16 + 255) / 256, 256, 0, stream>>>(xbf, out_w, out_b, out, Nn);
}
